// NovelDistanceLoss_50345606643883
// MI455X (gfx1250) — matrix operand order repaired and checked by reading, not compiled, not run on hardware
//
#include <hip/hip_runtime.h>

typedef __attribute__((ext_vector_type(16))) __bf16 v16bf;
typedef __attribute__((ext_vector_type(8)))  __bf16 v8bf;
typedef __attribute__((ext_vector_type(8)))  float  v8f;
typedef __attribute__((ext_vector_type(4)))  float  v4f;
typedef __attribute__((ext_vector_type(4)))  unsigned int u32x4;
typedef __attribute__((ext_vector_type(8)))  int    i32x8;
typedef __attribute__((ext_vector_type(4)))  int    i32x4;

#define DC 512
#define NR 1024
#define BZ 65536
#define KFRAGS (DC / 32)      // 16 K-fragments of 32
#define ROWS_PER_BLOCK 128    // 8 waves x 16 rows
#define NCHUNKS (NR / 16)     // 64 B-chunks of 16 rel rows
#define ROWSTRIDE_H 520       // 512 bf16 + 16B TDM pad -> 1040B = 260 dwords (4-bank shift/row)
#define MARGIN_F 1.0f

// ---------------------------------------------------------------------------
// Kernel 1: L2-normalize rel_weight rows, store BF16 copy in workspace.
// ---------------------------------------------------------------------------
__global__ __launch_bounds__(128) void rel_norm_kernel(const float* __restrict__ rel,
                                                       unsigned short* __restrict__ relb_)
{
    __bf16* relb = (__bf16*)relb_;
    const int row = blockIdx.x;
    const float* r = rel + (size_t)row * DC;

    float s = 0.f;
    for (int i = threadIdx.x; i < DC; i += 128) { float v = r[i]; s += v * v; }
#pragma unroll
    for (int m = 1; m < 32; m <<= 1) s += __shfl_xor(s, m, 32);

    __shared__ float red[4];
    if ((threadIdx.x & 31) == 0) red[threadIdx.x >> 5] = s;
    __syncthreads();
    const float tot = red[0] + red[1] + red[2] + red[3];
    const float scale = 1.0f / fmaxf(sqrtf(tot), 1e-12f);

    for (int i = threadIdx.x; i < DC; i += 128)
        relb[(size_t)row * DC + i] = (__bf16)(r[i] * scale);
}

// ---------------------------------------------------------------------------
// TDM: async-load one 16-row x 512-col bf16 tile of rel into LDS.
// D# per CDNA5 ISA ch.8: group0 = {count/type, lds_addr, global_addr},
// group1 = {data_size=2B, pad(16B per 1024B row), dims/strides}, groups 2/3 = 0.
// ---------------------------------------------------------------------------
__device__ __forceinline__ void tdm_load_chunk(const unsigned short* gsrc, unsigned ldsOff)
{
    unsigned long long ga = (unsigned long long)(uintptr_t)gsrc;
    u32x4 g0;
    g0[0] = 1u;                                               // count=1, no gather
    g0[1] = ldsOff;                                           // lds_addr (bytes)
    g0[2] = (unsigned)(ga & 0xFFFFFFFFull);                   // global_addr[31:0]
    g0[3] = (unsigned)((ga >> 32) & 0x01FFFFFFull) | (2u << 30); // ga[56:32] | type=2
    i32x8 g1;
    g1[0] = (1 << 16)            // data_size = 1 -> 2 bytes
          | (1 << 20)            // pad_enable
          | (7 << 22)            // pad_interval = 7 -> every 256 dwords (1024B = one row)
          | (3 << 25);           // pad_amount  = 3 -> 4 dwords (16B)
    g1[1] = (DC << 16);          // tensor_dim0[15:0] = 512
    g1[2] = (NR << 16);          // tensor_dim1[15:0] = 1024
    g1[3] = (DC << 16);          // tile_dim0 = 512
    g1[4] = 16;                  // tile_dim1 = 16 rows
    g1[5] = DC;                  // tensor_dim0_stride = 512
    g1[6] = 0;
    g1[7] = 0;
    i32x4 g2 = {0, 0, 0, 0};
    i32x4 g3 = {0, 0, 0, 0};
#if __clang_major__ >= 23
    i32x8 gpad = {0, 0, 0, 0, 0, 0, 0, 0};
    __builtin_amdgcn_tensor_load_to_lds(g0, g1, g2, g3, gpad, 0);
#else
    __builtin_amdgcn_tensor_load_to_lds(g0, g1, g2, g3, 0);
#endif
}

// ---------------------------------------------------------------------------
// Kernel 2: fused normalize(wo) -> BF16 WMMA GEMM vs rel (TDM-staged in LDS)
// -> distance -> masked-min / label gather -> hinge -> per-block partial sum.
// ---------------------------------------------------------------------------
__global__ __launch_bounds__(256) void dist_loss_kernel(const float* __restrict__ wo,
                                                        const int* __restrict__ in_y,
                                                        const unsigned short* __restrict__ relb_,
                                                        float* __restrict__ partials)
{
    __shared__ __align__(16) unsigned short ldsB[2][16 * ROWSTRIDE_H]; // 2 x 16.25KB
    __shared__ float wsum[8];

    const int wave = threadIdx.x >> 5;
    const int lane = threadIdx.x & 31;
    const int h = lane >> 4;   // wave half (selects K sub-pattern)
    const int l = lane & 15;   // row (A) / column (B/C) within tile
    const int rowBase = blockIdx.x * ROWS_PER_BLOCK + wave * 16;

    const float* wrow = wo + (size_t)(rowBase + l) * DC;

    // ---- kick off the B pipeline: chunk 0 into buffer 0 ----
    if (wave == 0)
        tdm_load_chunk(relb_, (unsigned)(uintptr_t)&ldsB[0][0]);

    // ---- row L2 norm: lane (h,l) sums half h of row l, combine via shfl ----
    float s = 0.f;
    {
        const v4f* p = (const v4f*)(wrow + h * (DC / 2));
#pragma unroll 4
        for (int i = 0; i < (DC / 2) / 4; ++i) {
            v4f x = p[i];
            s += x[0] * x[0] + x[1] * x[1] + x[2] * x[2] + x[3] * x[3];
        }
    }
    s += __shfl_xor(s, 16, 32);
    const float scale = 1.0f / fmaxf(sqrtf(s), 1e-12f);

    // ---- build resident A fragments (normalized, bf16) ----
    // 16-bit A 16x32 layout: lane half h, row l holds
    //   elements [0..7]  = K = 32f + 8h + (0..7)
    //   elements [8..15] = K = 32f + 8h + 16 + (0..7)
    v16bf afrag[KFRAGS];
#pragma unroll
    for (int f = 0; f < KFRAGS; ++f) {
        const int base = 32 * f + 8 * h;
        v4f x0 = *(const v4f*)(wrow + base);
        v4f x1 = *(const v4f*)(wrow + base + 4);
        v4f x2 = *(const v4f*)(wrow + base + 16);
        v4f x3 = *(const v4f*)(wrow + base + 20);
        v16bf a;
#pragma unroll
        for (int i = 0; i < 4; ++i) {
            a[i]      = (__bf16)(x0[i] * scale);
            a[4 + i]  = (__bf16)(x1[i] * scale);
            a[8 + i]  = (__bf16)(x2[i] * scale);
            a[12 + i] = (__bf16)(x3[i] * scale);
        }
        afrag[f] = a;
    }

    // ---- labels for the 8 C-rows this lane sees (C layout: M = r + 8h) ----
    int yv[8];
#pragma unroll
    for (int r = 0; r < 8; ++r) yv[r] = in_y[rowBase + 8 * h + r];

    float minv[8], posv[8];
#pragma unroll
    for (int r = 0; r < 8; ++r) { minv[r] = 3.0e38f; posv[r] = 0.f; }

    // ---- sweep all 64 B-chunks: TDM double-buffer, 16 WMMAs per chunk ----
    for (int c = 0; c < NCHUNKS; ++c) {
        if (wave == 0) {
            if (c + 1 < NCHUNKS) {
                tdm_load_chunk(relb_ + (size_t)(c + 1) * 16 * DC,
                               (unsigned)(uintptr_t)&ldsB[(c + 1) & 1][0]);
                __builtin_amdgcn_s_wait_tensorcnt((short)1);  // chunk c landed
            } else {
                __builtin_amdgcn_s_wait_tensorcnt((short)0);  // last chunk landed
            }
        }
        __syncthreads();  // chunk c visible to all waves

        // 16-bit B 32x16 layout: lane half h, col l holds K = 32f + 8h + (0..7), 32f + 16 + 8h + (0..7)
        const __bf16* brow = (const __bf16*)&ldsB[c & 1][l * ROWSTRIDE_H + 8 * h];
        auto loadb = [&](int f) -> v16bf {
            v8bf b0 = *(const v8bf*)(brow + 32 * f);
            v8bf b1 = *(const v8bf*)(brow + 32 * f + 16);
            return __builtin_shufflevector(b0, b1, 0, 1, 2, 3, 4, 5, 6, 7,
                                           8, 9, 10, 11, 12, 13, 14, 15);
        };

        // Depth-2 software pipeline: keep 2 B load-pairs in flight so each
        // WMMA waits on dscnt<=4 instead of 0 (LDS latency hidden by XDL pipe).
        v8f acc = {0.f, 0.f, 0.f, 0.f, 0.f, 0.f, 0.f, 0.f};
        v16bf bv0 = loadb(0);
        v16bf bv1 = loadb(1);
#pragma unroll
        for (int f = 0; f < KFRAGS - 2; ++f) {
            v16bf bnext = loadb(f + 2);
            acc = __builtin_amdgcn_wmma_f32_16x16x32_bf16(
                false, afrag[f], false, bv0, (short)0, acc, false, false);
            bv0 = bv1;
            bv1 = bnext;
        }
        acc = __builtin_amdgcn_wmma_f32_16x16x32_bf16(
            false, afrag[KFRAGS - 2], false, bv0, (short)0, acc, false, false);
        acc = __builtin_amdgcn_wmma_f32_16x16x32_bf16(
            false, afrag[KFRAGS - 1], false, bv1, (short)0, acc, false, false);

        const int n = c * 16 + l;  // global column; C layout: N = l
#pragma unroll
        for (int r = 0; r < 8; ++r) {
            float d = sqrtf(fmaxf(2.0f - 2.0f * acc[r], 0.0f));
            bool isy = (n == yv[r]);
            if (isy) posv[r] = d;
            else     minv[r] = fminf(minv[r], d);
        }
        __syncthreads();  // everyone done reading buf[c&1] before it is refilled
    }

    // ---- combine the 16 lanes of each half (same rows, disjoint N subsets) ----
#pragma unroll
    for (int m = 1; m < 16; m <<= 1) {
#pragma unroll
        for (int r = 0; r < 8; ++r) {
            minv[r] = fminf(minv[r], __shfl_xor(minv[r], m, 32));
            posv[r] = fmaxf(posv[r], __shfl_xor(posv[r], m, 32));
        }
    }
    float sum = 0.f;
#pragma unroll
    for (int r = 0; r < 8; ++r) {
        float hinge = fminf(fmaxf(MARGIN_F - minv[r], 0.f), 9999.f);
        sum += posv[r] + hinge;
    }
    sum += __shfl_xor(sum, 16, 32);  // add the other half's 8 rows

    if (lane == 0) wsum[wave] = sum;
    __syncthreads();
    if (threadIdx.x == 0) {
        float t = 0.f;
#pragma unroll
        for (int w = 0; w < 8; ++w) t += wsum[w];
        partials[blockIdx.x] = t;
    }
}

// ---------------------------------------------------------------------------
// Kernel 3: deterministic tree reduction of 512 block partials -> mean loss.
// ---------------------------------------------------------------------------
__global__ __launch_bounds__(256) void final_reduce_kernel(const float* __restrict__ partials,
                                                           float* __restrict__ out)
{
    __shared__ float sm[256];
    sm[threadIdx.x] = partials[threadIdx.x] + partials[threadIdx.x + 256];
    __syncthreads();
    for (int st = 128; st > 0; st >>= 1) {
        if (threadIdx.x < st) sm[threadIdx.x] += sm[threadIdx.x + st];
        __syncthreads();
    }
    if (threadIdx.x == 0) out[0] = sm[0] * (1.0f / (float)BZ);
}

// ---------------------------------------------------------------------------
extern "C" void kernel_launch(void* const* d_in, const int* in_sizes, int n_in,
                              void* d_out, int out_size, void* d_ws, size_t ws_size,
                              hipStream_t stream)
{
    const float* wo  = (const float*)d_in[0];   // [65536, 512] f32
    const float* rel = (const float*)d_in[1];   // [1024, 512]  f32
    const int*   iny = (const int*)d_in[2];     // [65536]      int
    float* out = (float*)d_out;                 // scalar loss

    // workspace: [0, 1MB) bf16 normalized rel; then 512 f32 block partials
    unsigned short* relb = (unsigned short*)d_ws;
    float* partials = (float*)((char*)d_ws + (size_t)NR * DC * sizeof(unsigned short));

    rel_norm_kernel<<<NR, 128, 0, stream>>>(rel, relb);
    dist_loss_kernel<<<BZ / ROWS_PER_BLOCK, 256, 0, stream>>>(wo, iny, relb, partials);
    final_reduce_kernel<<<1, 256, 0, stream>>>(partials, out);

    (void)in_sizes; (void)n_in; (void)out_size; (void)ws_size;
}
